// SegmentedDMPNet_5643587027114
// MI455X (gfx1250) — matrix operand order repaired and checked by reading, not compiled, not run on hardware
//
#include <hip/hip_runtime.h>

#define USE_ASYNC 1

typedef __bf16 bf16;
typedef __attribute__((ext_vector_type(8)))  __bf16 v8bf;
typedef __attribute__((ext_vector_type(16))) __bf16 v16bf;
typedef __attribute__((ext_vector_type(8)))  float  v8f;

__device__ __forceinline__ float clamp01(float v){ return fminf(fmaxf(v, 0.0f), 1.0f); }

// -------------------- conv1 (1->10ch, 5x5) + bias + maxpool2 + relu --------------------
// x: [256,1,100,100]  -> out: [256,10,48,48] fp32
__global__ void conv1_pool_relu(const float* __restrict__ x,
                                const float* __restrict__ w,
                                const float* __restrict__ bias,
                                float* __restrict__ out)
{
    int p = blockIdx.x * 256 + threadIdx.x;          // spatial 0..2303
    if (p >= 48 * 48) return;
    int c = blockIdx.y;
    int b = blockIdx.z;
    int py = p / 48, px = p % 48;
    float wr[25];
#pragma unroll
    for (int i = 0; i < 25; i++) wr[i] = w[c * 25 + i];
    const float* in = x + (long)b * 100 * 100;
    int oy = py * 2, ox = px * 2;
    float win[36];
#pragma unroll
    for (int r = 0; r < 6; r++)
#pragma unroll
        for (int q = 0; q < 6; q++)
            win[r * 6 + q] = in[(oy + r) * 100 + ox + q];
    float s00 = 0.f, s01 = 0.f, s10 = 0.f, s11 = 0.f;
#pragma unroll
    for (int ky = 0; ky < 5; ky++)
#pragma unroll
        for (int kx = 0; kx < 5; kx++) {
            float ww = wr[ky * 5 + kx];
            s00 += win[ky * 6 + kx] * ww;
            s01 += win[ky * 6 + kx + 1] * ww;
            s10 += win[(ky + 1) * 6 + kx] * ww;
            s11 += win[(ky + 1) * 6 + kx + 1] * ww;
        }
    float m = fmaxf(fmaxf(s00, s01), fmaxf(s10, s11)) + bias[c];
    out[(((long)b * 10 + c) * 48 + py) * 48 + px] = fmaxf(m, 0.f);
}

// -------------------- conv2 (10->20ch) + bias + maxpool2 + relu -> flattened bf16 --------------------
// h1: [256,10,48,48] -> A0 bf16 [256, 9696] (cols 0..9679 = flatten(C,H,W))
__global__ void conv2_pool_relu_bf16(const float* __restrict__ h1,
                                     const float* __restrict__ w,
                                     const float* __restrict__ bias,
                                     bf16* __restrict__ a0)
{
    long gid = (long)blockIdx.x * 256 + threadIdx.x; // exactly 256*9680 threads
    int b = (int)(gid / 9680);
    int r = (int)(gid % 9680);
    int c = r / 484;
    int q = r % 484;
    int py = q / 22, px = q % 22;
    int oy = py * 2, ox = px * 2;
    float s00 = 0.f, s01 = 0.f, s10 = 0.f, s11 = 0.f;
    for (int ic = 0; ic < 10; ic++) {
        const float* in = h1 + (((long)b * 10 + ic) * 48) * 48;
        const float* wp = w + (c * 10 + ic) * 25;
        float win[36];
#pragma unroll
        for (int rr = 0; rr < 6; rr++)
#pragma unroll
            for (int cc = 0; cc < 6; cc++)
                win[rr * 6 + cc] = in[(oy + rr) * 48 + ox + cc];
#pragma unroll
        for (int ky = 0; ky < 5; ky++)
#pragma unroll
            for (int kx = 0; kx < 5; kx++) {
                float ww = wp[ky * 5 + kx];
                s00 += win[ky * 6 + kx] * ww;
                s01 += win[ky * 6 + kx + 1] * ww;
                s10 += win[(ky + 1) * 6 + kx] * ww;
                s11 += win[(ky + 1) * 6 + kx + 1] * ww;
            }
    }
    float m = fmaxf(fmaxf(s00, s01), fmaxf(s10, s11)) + bias[c];
    a0[(long)b * 9696 + r] = (bf16)fmaxf(m, 0.f);
}

__global__ void pad_a0(bf16* __restrict__ a0)
{
    int i = blockIdx.x * 256 + threadIdx.x;          // 256*16 threads
    if (i < 256 * 16) {
        int b = i >> 4, j = i & 15;
        a0[(long)b * 9696 + 9680 + j] = (bf16)0.0f;
    }
}

// -------------------- transpose + f32->bf16 weight conversion --------------------
// W: [K][N] f32 -> Wt: [N][Kpad] bf16 (zeros for k in [K, Kpad)).
// grid: (Kpad/32, N/32), block 256 (32x8), LDS-tiled transpose.
__global__ __launch_bounds__(256)
void cvt_transpose_bf16(const float* __restrict__ W, bf16* __restrict__ Wt,
                        int K, int N, int Kpad)
{
    __shared__ float t[32][33];
    int tx = threadIdx.x & 31, ty = threadIdx.x >> 5;  // ty 0..7
    int kBase = blockIdx.x * 32, nBase = blockIdx.y * 32;
#pragma unroll
    for (int r = 0; r < 4; r++) {
        int k = kBase + ty + r * 8;
        t[ty + r * 8][tx] = (k < K) ? W[(long)k * N + nBase + tx] : 0.0f;
    }
    __syncthreads();
#pragma unroll
    for (int r = 0; r < 4; r++) {
        int n = nBase + ty + r * 8;
        Wt[(long)n * Kpad + kBase + tx] = (bf16)t[tx][ty + r * 8];
    }
}

// W2 [1024,142] -> W2t bf16 [144,1024] (transposed, rows 142..143 zero)
__global__ void cvt_t_w2(const float* __restrict__ W2, bf16* __restrict__ W2t)
{
    const int total = 144 * 1024;
    for (int i = blockIdx.x * blockDim.x + threadIdx.x; i < total;
         i += gridDim.x * blockDim.x) {
        int n = i >> 10, k = i & 1023;
        W2t[i] = (n < 142) ? (bf16)W2[k * 142 + n] : (bf16)0.0f;
    }
}

// -------------------- tiled WMMA GEMM: C = act(A @ Bt^T + bias) --------------------
// A:  [M][K]  bf16 row-major (K-contiguous)
// Bt: [N][K]  bf16 row-major (K-contiguous, pre-transposed weights)
// C:  [M][N]  bf16
// BM=BN=64, BK=32, 4 waves/block; wave computes 32x32 (2x2 tiles of 16x16).
// Double-buffered LDS filled with async global->LDS b128 copies (ASYNCcnt).
__global__ __launch_bounds__(128)
void gemm_bf16_wmma(const bf16* __restrict__ A, const bf16* __restrict__ Bt,
                    const float* __restrict__ bias, bf16* __restrict__ C,
                    int M, int N, int K, int doTanh)
{
    __shared__ __align__(16) bf16 lds_a[2][64 * 40];  // [buf][row][k] stride 40 halfs
    __shared__ __align__(16) bf16 lds_b[2][64 * 40];  // [buf][n][k]

    const int tid  = threadIdx.x;
    const int lane = tid & 31;
    const int wave = tid >> 5;
    const int wr = (wave & 1) * 32;    // wave row offset within 64
    const int wc = (wave >> 1) * 32;   // wave col offset within 64
    const int rowBlock = blockIdx.y * 64;
    const int colBlock = blockIdx.x * 64;
    const int m16 = lane & 15;
    const int kh  = lane >> 4;

    // Fill: each tile is 64 rows x 32 halfs = 256 chunks of 16B; thread does chunks tid, tid+128.
    const int fr = tid >> 2;           // base row (0..31)
    const int fc = (tid & 3) * 8;      // half-offset within row (0,8,16,24)

    v8f acc[2][2];
    v8f zero = {};
#pragma unroll
    for (int i = 0; i < 2; i++)
#pragma unroll
        for (int j = 0; j < 2; j++) acc[i][j] = zero;

    auto issue_tile = [&](int k0, int buf) {
#pragma unroll
        for (int i = 0; i < 2; i++) {
            int row = fr + i * 32;
            const bf16* ga = A  + (long)(rowBlock + row) * K + k0 + fc;
            const bf16* gb = Bt + (long)(colBlock + row) * K + k0 + fc;
#if USE_ASYNC
            unsigned oa = (unsigned)(unsigned long long)&lds_a[buf][row * 40 + fc];
            unsigned ob = (unsigned)(unsigned long long)&lds_b[buf][row * 40 + fc];
            asm volatile("global_load_async_to_lds_b128 %0, %1, off"
                         :: "v"(oa), "v"(ga) : "memory");
            asm volatile("global_load_async_to_lds_b128 %0, %1, off"
                         :: "v"(ob), "v"(gb) : "memory");
#else
            *(v8bf*)&lds_a[buf][row * 40 + fc] = *(const v8bf*)ga;
            *(v8bf*)&lds_b[buf][row * 40 + fc] = *(const v8bf*)gb;
#endif
        }
    };

    issue_tile(0, 0);
#if USE_ASYNC
    asm volatile("s_wait_asynccnt 0x0" ::: "memory");
#endif
    __syncthreads();

    const int nk = K >> 5;
    for (int it = 0; it < nk; ++it) {
        const int buf = it & 1;
        if (it + 1 < nk) issue_tile((it + 1) << 5, buf ^ 1);

        // A fragments: lane m=l%16 row m; elems 0..7 = K[kh*8..+7], 8..15 = K[16+kh*8..+7]
        v16bf af[2];
#pragma unroll
        for (int rt = 0; rt < 2; rt++) {
            int row = wr + rt * 16 + m16;
            v8bf lo = *(const v8bf*)&lds_a[buf][row * 40 + kh * 8];
            v8bf hi = *(const v8bf*)&lds_a[buf][row * 40 + 16 + kh * 8];
            af[rt] = __builtin_shufflevector(lo, hi, 0,1,2,3,4,5,6,7,8,9,10,11,12,13,14,15);
        }
        // B fragments: lane n=l%16 col n; elems 0..7 = K[kh*8..+7], 8..15 = K[16+kh*8..+7]
        v16bf bfv[2];
#pragma unroll
        for (int ct = 0; ct < 2; ct++) {
            int nn = wc + ct * 16 + m16;
            v8bf lo = *(const v8bf*)&lds_b[buf][nn * 40 + kh * 8];
            v8bf hi = *(const v8bf*)&lds_b[buf][nn * 40 + 16 + kh * 8];
            bfv[ct] = __builtin_shufflevector(lo, hi, 0,1,2,3,4,5,6,7,8,9,10,11,12,13,14,15);
        }
#pragma unroll
        for (int rt = 0; rt < 2; rt++)
#pragma unroll
            for (int ct = 0; ct < 2; ct++)
                acc[rt][ct] = __builtin_amdgcn_wmma_f32_16x16x32_bf16(
                    false, af[rt], false, bfv[ct], (short)0, acc[rt][ct], false, false);

        if (it + 1 < nk) {
#if USE_ASYNC
            asm volatile("s_wait_asynccnt 0x0" ::: "memory");
#endif
            __syncthreads();
        }
    }

    // store: C/D layout: VGPR r -> M = r + 8*(lane/16), N = lane%16
#pragma unroll
    for (int rt = 0; rt < 2; rt++)
#pragma unroll
        for (int ct = 0; ct < 2; ct++) {
            int col = colBlock + wc + ct * 16 + m16;
            float bv = bias[col];
#pragma unroll
            for (int r = 0; r < 8; r++) {
                int mrow = rowBlock + wr + rt * 16 + r + 8 * kh;
                float v = acc[rt][ct][r] + bv;
                if (doTanh) v = tanhf(v);
                C[(long)mrow * N + col] = (bf16)v;
            }
        }
}

// -------------------- FC2: one wave per 16x16 tile, B pre-transposed --------------------
// A: [256,1024] bf16; Bt: [144,1024] bf16; out: [256,144] fp32 (+bias on cols<142)
__global__ __launch_bounds__(32)
void fc2_wmma(const bf16* __restrict__ A, const bf16* __restrict__ Bt,
              const float* __restrict__ bias, float* __restrict__ out)
{
    const int lane  = threadIdx.x & 31;
    const int nBase = blockIdx.x * 16;   // 0..8 tiles
    const int mBase = blockIdx.y * 16;   // 0..15 tiles
    const int m16 = lane & 15;
    const int kh  = lane >> 4;
    v8f acc = {};
    for (int k0 = 0; k0 < 1024; k0 += 32) {
        const bf16* ap = A + (long)(mBase + m16) * 1024 + k0;
        v8bf alo = *(const v8bf*)(ap + kh * 8);
        v8bf ahi = *(const v8bf*)(ap + 16 + kh * 8);
        v16bf af = __builtin_shufflevector(alo, ahi, 0,1,2,3,4,5,6,7,8,9,10,11,12,13,14,15);
        const bf16* bp = Bt + (long)(nBase + m16) * 1024 + k0;
        v8bf blo = *(const v8bf*)(bp + kh * 8);
        v8bf bhi = *(const v8bf*)(bp + 16 + kh * 8);
        v16bf bfm = __builtin_shufflevector(blo, bhi, 0,1,2,3,4,5,6,7,8,9,10,11,12,13,14,15);
        acc = __builtin_amdgcn_wmma_f32_16x16x32_bf16(false, af, false, bfm, (short)0, acc, false, false);
    }
    int col = nBase + m16;
    float bv = (col < 142) ? bias[col] : 0.0f;
#pragma unroll
    for (int r = 0; r < 8; r++) {
        int mrow = mBase + r + 8 * kh;
        out[mrow * 144 + col] = acc[r] + bv;
    }
}

// -------------------- DMP integration --------------------
// o2: [256,144] fp32 (cols 0..141 valid); yt: [256,1000,2] fp32 clipped
__global__ void dmp_integrate(const float* __restrict__ o2, float* __restrict__ yt)
{
    int b = blockIdx.x;
    int l = threadIdx.x;
    if (l >= 20) return;
    int s = l >> 1, d = l & 1;
    const float* o = o2 + (long)b * 144;
    float y0   = clamp01(o[s * 2 + d]);
    float goal = clamp01(o[(s + 1) * 2 + d]);
    float w[6], c[6], hh[6];
#pragma unroll
    for (int n = 0; n < 6; n++) {
        w[n]  = o[22 + l * 6 + n];              // weights[s][d][n]
        c[n]  = __expf(-0.2f * (float)n);       // exp(-linspace(0,1,6))
        hh[n] = 14.696938456699069f / c[n];     // 6^1.5 / c
    }
    float xc = 1.0f, y = y0, dy = 0.0f;
    float* dst = yt + (long)b * 2000 + s * 200 + d;
    for (int t = 0; t < 100; t++) {
        xc *= 0.99f;                            // xc -= CS_AX*xc*TAU*DT
        float psum = 0.f, wsum = 0.f;
#pragma unroll
        for (int n = 0; n < 6; n++) {
            float dd = xc - c[n];
            float p = __expf(-hh[n] * dd * dd);
            psum += p;
            wsum += w[n] * p;
        }
        float f   = xc * (goal - y0) * wsum / psum;
        float ddy = 25.0f * (6.25f * (goal - y) - dy) + f;
        dy += ddy * 0.01f;
        y  += dy * 0.01f;
        dst[t * 2] = clamp01(y);
    }
}

// -------------------- launcher --------------------
extern "C" void kernel_launch(void* const* d_in, const int* in_sizes, int n_in,
                              void* d_out, int out_size, void* d_ws, size_t ws_size,
                              hipStream_t stream)
{
    (void)in_sizes; (void)n_in; (void)out_size; (void)ws_size;
    const float* x   = (const float*)d_in[0];
    const float* c1w = (const float*)d_in[1];
    const float* c1b = (const float*)d_in[2];
    const float* c2w = (const float*)d_in[3];
    const float* c2b = (const float*)d_in[4];
    const float* W0  = (const float*)d_in[5];
    const float* b0  = (const float*)d_in[6];
    const float* W1  = (const float*)d_in[7];
    const float* b1  = (const float*)d_in[8];
    const float* W2  = (const float*)d_in[9];
    const float* b2  = (const float*)d_in[10];
    float* out = (float*)d_out;

    char* ws = (char*)d_ws;
    auto carve = [&](size_t bytes) -> char* {
        char* p = ws;
        ws += (bytes + 255) & ~(size_t)255;
        return p;
    };
    float* h1  = (float*)carve((size_t)256 * 10 * 48 * 48 * 4); // 23.6 MB
    bf16*  A0  = (bf16*) carve((size_t)256 * 9696 * 2);         //  5.0 MB
    bf16*  W0t = (bf16*) carve((size_t)2048 * 9696 * 2);        // 39.7 MB  [N][Kpad]
    bf16*  W1t = (bf16*) carve((size_t)1024 * 2048 * 2);        //  4.2 MB  [N][K]
    bf16*  W2t = (bf16*) carve((size_t)144 * 1024 * 2);         //  0.3 MB
    bf16*  A1  = (bf16*) carve((size_t)256 * 2048 * 2);         //  1.0 MB
    bf16*  A2  = (bf16*) carve((size_t)256 * 1024 * 2);         //  0.5 MB
    float* O2  = (float*)carve((size_t)256 * 144 * 4);          //  0.15 MB

    conv1_pool_relu<<<dim3(9, 10, 256), 256, 0, stream>>>(x, c1w, c1b, h1);
    conv2_pool_relu_bf16<<<dim3(9680), 256, 0, stream>>>(h1, c2w, c2b, A0);
    pad_a0<<<16, 256, 0, stream>>>(A0);
    // weight transposes: W0 [9680,2048] -> W0t [2048,9696]; W1 [2048,1024] -> W1t [1024,2048]
    cvt_transpose_bf16<<<dim3(9696 / 32, 2048 / 32), 256, 0, stream>>>(W0, W0t, 9680, 2048, 9696);
    cvt_transpose_bf16<<<dim3(2048 / 32, 1024 / 32), 256, 0, stream>>>(W1, W1t, 2048, 1024, 2048);
    cvt_t_w2<<<144, 256, 0, stream>>>(W2, W2t);

    // FC0: [256,9696] x W0t^T (+b0, tanh) -> A1 [256,2048]
    gemm_bf16_wmma<<<dim3(2048 / 64, 256 / 64), 128, 0, stream>>>(A0, W0t, b0, A1, 256, 2048, 9696, 1);
    // FC1: [256,2048] x W1t^T (+b1, tanh) -> A2 [256,1024]
    gemm_bf16_wmma<<<dim3(1024 / 64, 256 / 64), 128, 0, stream>>>(A1, W1t, b1, A2, 256, 1024, 2048, 1);
    // FC2: [256,1024] x W2t^T (+b2) -> O2 fp32 [256,144]
    fc2_wmma<<<dim3(9, 16), 32, 0, stream>>>(A2, W2t, b2, O2);

    dmp_integrate<<<256, 32, 0, stream>>>(O2, out);
}
